// NeuralODEModel_19121194402275
// MI455X (gfx1250) — matrix operand order repaired and checked by reading, not compiled, not run on hardware
//
#include <hip/hip_runtime.h>
#include <hip/hip_bf16.h>

// ---------------------------------------------------------------------------
// NeuralODE on MI455X (gfx1250): 183-GEMM chain, compute bound.
// bf16 WMMA (v_wmma_f32_16x16x32_bf16) with f32 accumulation.
// Intermediate activations held in bf16 (quantization point unchanged);
// register-prefetch pipeline overlaps next K-block globals with WMMA work.
// ---------------------------------------------------------------------------

typedef __attribute__((ext_vector_type(16))) __bf16 v16bf;
typedef __attribute__((ext_vector_type(8)))  float  v8f;

#define BK 32
#define SA_STRIDE 40   // 32 + 8 pad (bf16 elements) to break bank conflicts
#define SB_STRIDE 40

static __device__ __forceinline__ unsigned short f2bf(float f) {
  unsigned u; __builtin_memcpy(&u, &f, 4);
  u += 0x7FFFu + ((u >> 16) & 1u);   // round-to-nearest-even
  return (unsigned short)(u >> 16);
}

static __device__ __forceinline__ unsigned packbf(float a, float b) {
  return (unsigned)f2bf(a) | ((unsigned)f2bf(b) << 16);
}

static __device__ __forceinline__ float gelu_exact(float x) {
  return 0.5f * x * (1.0f + erff(x * 0.70710678118654752f));
}

// C[M,N] = act(A[M,K](row stride lda) @ Wt[N,K](bf16)^T + bias[N])
// A is f32 (ABF=false) or bf16 (ABF=true); C is f32 (OBF=false) or bf16.
// Block: 256 threads = 8 waves = WM x WN wave grid; wave tile = MI*16 x NI*16.
template <int MI, int NI, int WM, int WN, bool ABF, bool OBF>
__global__ __launch_bounds__(256) void gemm_bias_act(
    const void* __restrict__ Av, int lda,
    const unsigned short* __restrict__ Wt,   // [N,K] bf16 (pre-transposed)
    const float* __restrict__ bias,
    void* __restrict__ Cv, int N, int K, int act)
{
  static_assert(WM * WN == 8, "8 waves per block");
  constexpr int BMt  = WM * MI * 16;
  constexpr int BNt  = WN * NI * 16;
  constexpr int AREGF = (BMt * BK) / (256 * 4);  // float4/thread (f32 A)
  constexpr int AREGH = (BMt * BK) / (256 * 8);  // uint4/thread  (bf16 A)
  constexpr int BREG  = (BNt * BK) / (256 * 8);  // uint4/thread  (bf16 B)

  __shared__ __align__(16) unsigned short sA[BMt * SA_STRIDE];
  __shared__ __align__(16) unsigned short sB[BNt * SB_STRIDE];

  const int t    = threadIdx.x;
  const int lane = t & 31;
  const int wave = t >> 5;
  const int wm   = wave / WN;
  const int wn   = wave % WN;
  const int half = lane >> 4;   // 0: lanes 0-15, 1: lanes 16-31
  const int lm   = lane & 15;

  const int m0 = blockIdx.y * BMt;
  const int n0 = blockIdx.x * BNt;

  const float*          Af = (const float*)Av;
  const unsigned short* Ah = (const unsigned short*)Av;

  v8f acc[MI][NI] = {};
  float4 aRegF[ABF ? 1 : AREGF];
  uint4  aRegH[ABF ? AREGH : 1];
  uint4  bReg[BREG];

  auto loadTile = [&](int kb) {
    if constexpr (ABF) {
      #pragma unroll
      for (int r = 0; r < AREGH; ++r) {
        int p = (t + r * 256) * 8;          // flat bf16 index in BM x 32 tile
        int m = p >> 5, k = p & 31;
        aRegH[r] = *(const uint4*)(Ah + (size_t)(m0 + m) * lda + kb + k);
      }
    } else {
      #pragma unroll
      for (int r = 0; r < AREGF; ++r) {
        int p = (t + r * 256) * 4;          // flat f32 index in BM x 32 tile
        int m = p >> 5, k = p & 31;
        aRegF[r] = *(const float4*)(Af + (size_t)(m0 + m) * lda + kb + k);
      }
    }
    #pragma unroll
    for (int r = 0; r < BREG; ++r) {
      int p = (t + r * 256) * 8;            // flat bf16 index in BN x 32 tile
      int n = p >> 5, k = p & 31;
      bReg[r] = *(const uint4*)(Wt + (size_t)(n0 + n) * K + kb + k);
    }
  };

  loadTile(0);

  for (int kb = 0; kb < K; kb += BK) {
    __syncthreads();
    // ---- commit prefetched tile to LDS (f32 A converts to bf16) ----
    if constexpr (ABF) {
      #pragma unroll
      for (int r = 0; r < AREGH; ++r) {
        int p = (t + r * 256) * 8;
        int m = p >> 5, k = p & 31;
        *(uint4*)&sA[m * SA_STRIDE + k] = aRegH[r];
      }
    } else {
      #pragma unroll
      for (int r = 0; r < AREGF; ++r) {
        int p = (t + r * 256) * 4;
        int m = p >> 5, k = p & 31;
        uint2 pk; pk.x = packbf(aRegF[r].x, aRegF[r].y);
                  pk.y = packbf(aRegF[r].z, aRegF[r].w);
        *(uint2*)&sA[m * SA_STRIDE + k] = pk;
      }
    }
    #pragma unroll
    for (int r = 0; r < BREG; ++r) {
      int p = (t + r * 256) * 8;
      int n = p >> 5, k = p & 31;
      *(uint4*)&sB[n * SB_STRIDE + k] = bReg[r];
    }
    __syncthreads();

    // ---- issue next K-block global loads; wait lands at next LDS commit ----
    if (kb + BK < K) loadTile(kb + BK);

    // ---- fragments per ISA 16-bit A(16x32)/B(32x16) layouts ----
    v16bf afrag[MI], bfrag[NI];
    #pragma unroll
    for (int mi = 0; mi < MI; ++mi) {
      // lanes 0-15: K 0..7 (v0-3) + 16..23 (v4-7); lanes 16-31: K 8..15 + 24..31
      int row = (wm * MI + mi) * 16 + lm;
      const unsigned short* src = &sA[row * SA_STRIDE + half * 8];
      union { uint4 u[2]; v16bf v; } fa;
      fa.u[0] = *(const uint4*)(src);        // K c..c+7
      fa.u[1] = *(const uint4*)(src + 16);   // K c+16..c+23
      afrag[mi] = fa.v;
    }
    #pragma unroll
    for (int ni = 0; ni < NI; ++ni) {
      // lanes 0-15: K 0..7 (v0-3) + 16..23 (v4-7); lanes 16-31: K 8..15 + 24..31
      int col = (wn * NI + ni) * 16 + lm;
      const unsigned short* src = &sB[col * SB_STRIDE + half * 8];
      union { uint4 u[2]; v16bf v; } fb;
      fb.u[0] = *(const uint4*)(src);
      fb.u[1] = *(const uint4*)(src + 16);
      bfrag[ni] = fb.v;
    }

    #pragma unroll
    for (int mi = 0; mi < MI; ++mi)
      #pragma unroll
      for (int ni = 0; ni < NI; ++ni)
        acc[mi][ni] = __builtin_amdgcn_wmma_f32_16x16x32_bf16(
            false, afrag[mi], false, bfrag[ni],
            (short)0, acc[mi][ni], false, false);
  }

  // ---- epilogue: bias + optional exact GELU; C/D: vgpr r -> row half*8+r ----
  float*          Cf = (float*)Cv;
  unsigned short* Ch = (unsigned short*)Cv;
  #pragma unroll
  for (int ni = 0; ni < NI; ++ni) {
    int col = n0 + (wn * NI + ni) * 16 + lm;
    float bv = bias[col];
    #pragma unroll
    for (int mi = 0; mi < MI; ++mi) {
      #pragma unroll
      for (int r = 0; r < 8; ++r) {
        int row = m0 + (wm * MI + mi) * 16 + half * 8 + r;
        float v = acc[mi][ni][r] + bv;
        if (act) v = gelu_exact(v);
        if constexpr (OBF) Ch[(size_t)row * N + col] = f2bf(v);
        else               Cf[(size_t)row * N + col] = v;
      }
    }
  }
}

// Wt[n*K + k] = bf16(W[k*N + n]) ; coalesced both sides via LDS tile
__global__ __launch_bounds__(256) void convert_transpose_bf16(
    const float* __restrict__ W, unsigned short* __restrict__ Wt, int K, int N)
{
  __shared__ float tile[32][33];
  int tx = threadIdx.x & 31;
  int ty = threadIdx.x >> 5;      // 0..7
  int kbase = blockIdx.x * 32;
  int nbase = blockIdx.y * 32;
  #pragma unroll
  for (int i = 0; i < 4; ++i)
    tile[ty + i * 8][tx] = W[(size_t)(kbase + ty + i * 8) * N + nbase + tx];
  __syncthreads();
  #pragma unroll
  for (int i = 0; i < 4; ++i)
    Wt[(size_t)(nbase + ty + i * 8) * K + kbase + tx] = f2bf(tile[tx][ty + i * 8]);
}

// y = h + a*p + b*q + c*r + d*s   (float4 vectorized)
__global__ __launch_bounds__(256) void axpy4(
    float* __restrict__ y, const float* __restrict__ h,
    const float* __restrict__ p, float a, const float* __restrict__ q, float b,
    const float* __restrict__ r, float c, const float* __restrict__ s, float d,
    int n4)
{
  int i = blockIdx.x * 256 + threadIdx.x;
  if (i >= n4) return;
  float4 hv = ((const float4*)h)[i];
  float4 pv = ((const float4*)p)[i];
  float4 qv = ((const float4*)q)[i];
  float4 rv = ((const float4*)r)[i];
  float4 sv = ((const float4*)s)[i];
  float4 o;
  o.x = hv.x + a * pv.x + b * qv.x + c * rv.x + d * sv.x;
  o.y = hv.y + a * pv.y + b * qv.y + c * rv.y + d * sv.y;
  o.z = hv.z + a * pv.z + b * qv.z + c * rv.z + d * sv.z;
  o.w = hv.w + a * pv.w + b * qv.w + c * rv.w + d * sv.w;
  ((float4*)y)[i] = o;
}

// ---------------------------------------------------------------------------

extern "C" void kernel_launch(void* const* d_in, const int* in_sizes, int n_in,
                              void* d_out, int out_size, void* d_ws, size_t ws_size,
                              hipStream_t stream) {
  (void)in_sizes; (void)n_in; (void)out_size; (void)ws_size;
  enum { Bsz = 2048, Seq = 16, DIN = 512, H = 1024, DOUT = 64 };

  const float* x   = (const float*)d_in[0];
  const float* Wi  = (const float*)d_in[1];
  const float* bi  = (const float*)d_in[2];
  const float* W1  = (const float*)d_in[3];
  const float* b1  = (const float*)d_in[4];
  const float* W2  = (const float*)d_in[5];
  const float* b2  = (const float*)d_in[6];
  const float* W3  = (const float*)d_in[7];
  const float* b3  = (const float*)d_in[8];
  const float* Wo1 = (const float*)d_in[9];
  const float* bo1 = (const float*)d_in[10];
  const float* Wo2 = (const float*)d_in[11];
  const float* bo2 = (const float*)d_in[12];

  char* ws = (char*)d_ws;
  size_t off = 0;
  auto take = [&](size_t bytes) -> char* {
    char* p = ws + off; off += (bytes + 255) & ~(size_t)255; return p;
  };
  unsigned short* W1t  = (unsigned short*)take((size_t)H * H * 2);
  unsigned short* W2t  = (unsigned short*)take((size_t)H * H * 2);
  unsigned short* W3t  = (unsigned short*)take((size_t)H * H * 2);
  unsigned short* Wit  = (unsigned short*)take((size_t)H * DIN * 2);        // [H, DIN]
  unsigned short* Wo1t = (unsigned short*)take((size_t)(H / 2) * H * 2);    // [H/2, H]
  unsigned short* Wo2t = (unsigned short*)take((size_t)DOUT * (H / 2) * 2); // [DOUT, H/2]
  float* h   = (float*)take((size_t)Bsz * H * 4);
  float* tin = (float*)take((size_t)Bsz * H * 4);
  float* k1  = (float*)take((size_t)Bsz * H * 4);
  float* k2  = (float*)take((size_t)Bsz * H * 4);
  float* k3  = (float*)take((size_t)Bsz * H * 4);
  unsigned short* t1b    = (unsigned short*)take((size_t)Bsz * H * 2);       // bf16
  unsigned short* t2b    = (unsigned short*)take((size_t)Bsz * H * 2);       // bf16
  unsigned short* hhalfb = (unsigned short*)take((size_t)Bsz * (H / 2) * 2); // bf16

  // ---- weight prep: transpose + bf16 convert (L2-resident afterwards) ----
  convert_transpose_bf16<<<dim3(DIN / 32, H / 32),        256, 0, stream>>>(Wi,  Wit,  DIN,   H);
  convert_transpose_bf16<<<dim3(H / 32, H / 32),          256, 0, stream>>>(W1,  W1t,  H,     H);
  convert_transpose_bf16<<<dim3(H / 32, H / 32),          256, 0, stream>>>(W2,  W2t,  H,     H);
  convert_transpose_bf16<<<dim3(H / 32, H / 32),          256, 0, stream>>>(W3,  W3t,  H,     H);
  convert_transpose_bf16<<<dim3(H / 32, (H / 2) / 32),    256, 0, stream>>>(Wo1, Wo1t, H,     H / 2);
  convert_transpose_bf16<<<dim3((H / 2) / 32, DOUT / 32), 256, 0, stream>>>(Wo2, Wo2t, H / 2, DOUT);

  const dim3 gFull(H / 128, Bsz / 128);        // N=1024, 128x128 block tiles
  // h0 = x[:,0,:] @ Wi + bi  (row stride Seq*DIN selects s==0 slice)
  gemm_bias_act<4, 2, 2, 4, false, false>
      <<<gFull, 256, 0, stream>>>(x, Seq * DIN, Wit, bi, h, H, DIN, 0);

  const float dt = 1.0f / (float)(Seq - 1);
  const int n4 = Bsz * H / 4;
  const int axg = n4 / 256;

  // f(h) = gelu(gelu(gelu(h@W1+b1)@W2+b2)@W3+b3); middle layers stay bf16
  auto f3 = [&](const float* in, float* outbuf) {
    gemm_bias_act<4, 2, 2, 4, false, true>
        <<<gFull, 256, 0, stream>>>(in,  H, W1t, b1, t1b,    H, H, 1);
    gemm_bias_act<4, 2, 2, 4, true, true>
        <<<gFull, 256, 0, stream>>>(t1b, H, W2t, b2, t2b,    H, H, 1);
    gemm_bias_act<4, 2, 2, 4, true, false>
        <<<gFull, 256, 0, stream>>>(t2b, H, W3t, b3, outbuf, H, H, 1);
  };

  for (int step = 0; step < Seq - 1; ++step) {
    f3(h, k1);                                                          // k1 = f(h)
    axpy4<<<axg, 256, 0, stream>>>(tin, h, k1, dt / 3.0f, k1, 0.0f, k1, 0.0f, k1, 0.0f, n4);
    f3(tin, k2);                                                        // k2 = f(h + dt*k1/3)
    axpy4<<<axg, 256, 0, stream>>>(tin, h, k1, -dt / 3.0f, k2, dt, k1, 0.0f, k1, 0.0f, n4);
    f3(tin, k3);                                                        // k3 = f(h + dt*(k2 - k1/3))
    axpy4<<<axg, 256, 0, stream>>>(tin, h, k1, dt, k2, -dt, k3, dt, k1, 0.0f, n4);
    f3(tin, tin);                                                       // k4 -> tin (tin free after layer1)
    axpy4<<<axg, 256, 0, stream>>>(h, h, k1, dt / 8.0f, k2, 3.0f * dt / 8.0f,
                                   k3, 3.0f * dt / 8.0f, tin, dt / 8.0f, n4);
  }

  // out = gelu(h @ Wo1 + bo1) @ Wo2 + bo2
  gemm_bias_act<4, 2, 2, 4, false, true>
      <<<dim3((H / 2) / 128, Bsz / 128), 256, 0, stream>>>(h, H, Wo1t, bo1, hhalfb, H / 2, H, 1);
  gemm_bias_act<2, 2, 4, 2, true, false>
      <<<dim3(DOUT / 64, Bsz / 128), 256, 0, stream>>>(hhalfb, H / 2, Wo2t, bo2,
                                                       d_out, DOUT, H / 2, 0);
}
